// Scorer_22531398435242
// MI455X (gfx1250) — compile-verified, matrix operand order checked by reading, not run on hardware
//
#include <hip/hip_runtime.h>

// Cosine similarity: out[n][l] = dot(text[n], label[l]) / max(||text n||*||label l||, eps)
// text: [65536, 256] f32, labels: [1024, 256] f32, out: [65536, 1024] f32.
// f16 WMMA GEMM (f32 accum); f32 norms from original data; reciprocal-norm epilogue.

typedef __fp16   v2hp __attribute__((ext_vector_type(2)));   // cvt_pkrtz result type
typedef _Float16 v8h  __attribute__((ext_vector_type(8)));
typedef _Float16 v16h __attribute__((ext_vector_type(16)));
typedef float    v8f  __attribute__((ext_vector_type(8)));

#define KDIM     256
#define BM       128
#define BN       128
#define LDSK     264    // 256 + 8 halves pad -> 528B row stride, conflict-free b128 frag loads
#define INV_EPS  1e8f   // 1/eps; min(ra*rb, 1/eps) == 1/max(tn*ln, eps)

// Load one 16x32 f16 WMMA operand fragment from LDS (row-major, K contiguous).
// Lane L (0..15): K=[k0..k0+7] and [k0+16..k0+23] of row row0+L;
// lane L+16:      K=[k0+8..k0+15] and [k0+24..k0+31].
__device__ __forceinline__ v16h load_frag(const _Float16* lds, int row0, int k0, int lane) {
    const int l15 = lane & 15;
    const int hi  = (lane >> 4) & 1;
    const _Float16* p = lds + (size_t)(row0 + l15) * LDSK + k0 + (hi << 3);
    v8h c0 = *(const v8h*)(p);        // ds_load_b128
    v8h c1 = *(const v8h*)(p + 16);   // ds_load_b128
    return __builtin_shufflevector(c0, c1, 0,1,2,3,4,5,6,7,8,9,10,11,12,13,14,15);
}

// Stage a 128x256 f32 tile into LDS as f16 (packed converts), accumulating per-row
// sum-of-squares in f32. Flat-chunk mapping: consecutive threads read consecutive 32B.
__device__ __forceinline__ void fill_tile(const float* __restrict__ g, _Float16* lds,
                                          float* nrm, int tid) {
#pragma unroll
    for (int j = 0; j < (BM * (KDIM / 8)) / 256; ++j) {  // 16 chunks/thread
        const int c   = tid + j * 256;
        const int row = c >> 5;        // 32 chunks per 256-elem row
        const int kc  = c & 31;
        const float4* gp = (const float4*)(g + (size_t)c * 8);
        float4 f0 = gp[0];
        float4 f1 = gp[1];
        float ss = f0.x*f0.x + f0.y*f0.y + f0.z*f0.z + f0.w*f0.w
                 + f1.x*f1.x + f1.y*f1.y + f1.z*f1.z + f1.w*f1.w;
        atomicAdd(&nrm[row], ss);      // ds_add_f32
        union { v2hp h2[4]; v8h h8; } u;
        u.h2[0] = __builtin_amdgcn_cvt_pkrtz(f0.x, f0.y);  // v_cvt_pk_f16_f32
        u.h2[1] = __builtin_amdgcn_cvt_pkrtz(f0.z, f0.w);
        u.h2[2] = __builtin_amdgcn_cvt_pkrtz(f1.x, f1.y);
        u.h2[3] = __builtin_amdgcn_cvt_pkrtz(f1.z, f1.w);
        *(v8h*)(lds + (size_t)row * LDSK + kc * 8) = u.h8;  // ds_store_b128
    }
}

__global__ __launch_bounds__(256, 2)
void cosine_sim_wmma(const float* __restrict__ text,
                     const float* __restrict__ labels,
                     float* __restrict__ out,
                     int ldOut) {
    extern __shared__ char smem[];
    _Float16* lds_a  = (_Float16*)smem;                   // 128 x 264 halves
    _Float16* lds_b  = lds_a + (size_t)BM * LDSK;         // 128 x 264 halves
    float*    lds_an = (float*)(lds_b + (size_t)BN * LDSK);
    float*    lds_bn = lds_an + BM;

    const int tid   = threadIdx.x;
    const int lane  = tid & 31;
    const int wave  = tid >> 5;         // 8 waves
    const int waveM = wave >> 2;        // 0..1 -> 64 rows each
    const int waveN = wave & 3;         // 0..3 -> 32 cols each
    const int aRow0 = waveM * 64;
    const int bRow0 = waveN * 32;

    if (tid < BM) { lds_an[tid] = 0.0f; lds_bn[tid] = 0.0f; }
    __syncthreads();

    const float* aG = text   + (size_t)blockIdx.x * BM * KDIM;
    const float* bG = labels + (size_t)blockIdx.y * BN * KDIM;
    fill_tile(aG, lds_a, lds_an, tid);
    fill_tile(bG, lds_b, lds_bn, tid);
    __syncthreads();

    // Convert sum-of-squares -> reciprocal norms in place (nobody reads these
    // until after the barrier that follows the K loop).
    if (tid < BM)            lds_an[tid]      = __frsqrt_rn(lds_an[tid]);
    else if (tid < BM + BN)  lds_bn[tid - BM] = __frsqrt_rn(lds_bn[tid - BM]);

    v8f acc[4][2] = {};   // 4 M-frags x 2 N-frags per wave = 64x32 output

#pragma unroll
    for (int k0 = 0; k0 < KDIM; k0 += 32) {
        v16h aF[4], bF[2];
#pragma unroll
        for (int i = 0; i < 4; ++i) aF[i] = load_frag(lds_a, aRow0 + i * 16, k0, lane);
#pragma unroll
        for (int j = 0; j < 2; ++j) bF[j] = load_frag(lds_b, bRow0 + j * 16, k0, lane);
#pragma unroll
        for (int i = 0; i < 4; ++i)
#pragma unroll
            for (int j = 0; j < 2; ++j)
                acc[i][j] = __builtin_amdgcn_wmma_f32_16x16x32_f16(
                    /*neg_a=*/false, aF[i], /*neg_b=*/false, bF[j],
                    /*c_mod=*/(short)0, acc[i][j],
                    /*reuse_a=*/false, /*reuse_b=*/false);
    }

    __syncthreads();   // reciprocal norms visible to all waves

    // Epilogue: C-layout is VGPR r -> M = r (lanes 0-15) / r+8 (lanes 16-31), N = lane&15.
    // out = acc * min(ra*rb, 1/eps)  ==  acc / max(tn*ln, eps).
    // Non-temporal stores keep the 64MB A matrix resident in the 192MB L2.
    const int l15    = lane & 15;
    const int hiRows = (lane >> 4) << 3;
#pragma unroll
    for (int i = 0; i < 4; ++i) {
#pragma unroll
        for (int j = 0; j < 2; ++j) {
            const int nLoc = bRow0 + j * 16 + l15;
            const float rb = lds_bn[nLoc];
            const size_t col = (size_t)blockIdx.y * BN + nLoc;
#pragma unroll
            for (int r = 0; r < 8; ++r) {
                const int mLoc = aRow0 + i * 16 + hiRows + r;
                const float ra  = lds_an[mLoc];
                const float inv = fminf(ra * rb, INV_EPS);
                const size_t rowG = (size_t)blockIdx.x * BM + mLoc;
                __builtin_nontemporal_store(acc[i][j][r] * inv,
                                            &out[rowG * (size_t)ldOut + col]);
            }
        }
    }
}

extern "C" void kernel_launch(void* const* d_in, const int* in_sizes, int n_in,
                              void* d_out, int out_size, void* d_ws, size_t ws_size,
                              hipStream_t stream) {
    const float* text   = (const float*)d_in[0];   // [N, 256] f32
    const float* labels = (const float*)d_in[1];   // [L, 256] f32
    float* out = (float*)d_out;                    // [N, L] f32

    const int N = in_sizes[0] / KDIM;   // 65536
    const int L = in_sizes[1] / KDIM;   // 1024

    const size_t smemBytes = (size_t)(BM + BN) * LDSK * sizeof(_Float16)
                           + (size_t)(BM + BN) * sizeof(float);  // ~136 KB (2 WGs/WGP)

    (void)hipFuncSetAttribute((const void*)cosine_sim_wmma,
                              hipFuncAttributeMaxDynamicSharedMemorySize,
                              (int)smemBytes);

    dim3 grid(N / BM, L / BN);   // (512, 8); A re-reads across y hit L2 (A fits in 192MB)
    dim3 block(256);             // 8 wave32
    cosine_sim_wmma<<<grid, block, smemBytes, stream>>>(text, labels, out, L);
}
